// InfiniAttentionALiBi_27736898798184
// MI455X (gfx1250) — compile-verified, matrix operand order checked by reading, not run on hardware
//
#include <hip/hip_runtime.h>
#include <hip/hip_bf16.h>

typedef __bf16 bf16_t;
typedef __attribute__((ext_vector_type(16))) __bf16 v16bf;
typedef __attribute__((ext_vector_type(8)))  __bf16 v8bf;
typedef __attribute__((ext_vector_type(4)))  __bf16 v4bf;
typedef __attribute__((ext_vector_type(8)))  float  v8f;
typedef __attribute__((ext_vector_type(4)))  int    i32x4;

#define NHEAD 16
#define DHEAD 128
#define HID   2048
#define BATCH 4
#define SEQ   4096
#define MROWS (BATCH * SEQ)   // 16384

typedef __attribute__((address_space(3))) char lds_char;

// ---- CDNA5 async global->LDS path (guarded; falls back to sync copies) ----
#if defined(__gfx1250__) && __has_builtin(__builtin_amdgcn_global_load_async_to_lds_b128) && \
    __has_builtin(__builtin_amdgcn_s_wait_asynccnt)
#define USE_ASYNC 1
__device__ inline void async_b128(const void* g, void* lds) {
  __builtin_amdgcn_global_load_async_to_lds_b128((i32x4*)g, (i32x4*)lds, 0, 0);
}
__device__ inline void async_wait() { __builtin_amdgcn_s_wait_asynccnt(0); }
#else
#define USE_ASYNC 0
#endif

// ---- CDNA5 Tensor Data Mover path (guarded; clang-22=5 args, clang-23=6) ----
#if defined(__gfx1250__) && __has_builtin(__builtin_amdgcn_tensor_load_to_lds) && \
    __has_builtin(__builtin_amdgcn_s_wait_tensorcnt)
#define USE_TDM 1
typedef __attribute__((ext_vector_type(4))) unsigned int u32x4;
typedef __attribute__((ext_vector_type(8))) int i32x8;
#else
#define USE_TDM 0
#endif

__device__ inline float elu1(float x) { return x > 0.0f ? x + 1.0f : __expf(x); }

// 16-bit A/B fragment from an LDS row holding K=0..31 contiguously.
// ISA 7.12.2: lanes 0-15 take K {0-7,16-23}, lanes 16-31 take K {8-15,24-31}.
__device__ inline v16bf frag_row(const bf16_t* rowp, int half) {
  const v8bf lo = *(const v8bf*)(rowp + half * 8);
  const v8bf hi = *(const v8bf*)(rowp + 16 + half * 8);
  return __builtin_shufflevector(lo, hi, 0, 1, 2, 3, 4, 5, 6, 7,
                                 8, 9, 10, 11, 12, 13, 14, 15);
}

// ---------------------------------------------------------------------------
// C[M,N] = X[M,K] * W[N,K]^T, bf16 -> f32 accum via WMMA.
// BM=128, BN=256, BK=64; 8 waves in a 2x4 grid; each wave owns 64x64
// (16 x v8f accumulators -> 16 WMMA per 8 fragment loads per K-step).
// MODE 0: bf16 elu+1 (sigma_q)   MODE 1: bf16 (elu+1)*alibi_w (wk)
// MODE 2: bf16 raw (v)           MODE 3: f32 raw (output projection)
// ---------------------------------------------------------------------------
template <int MODE>
__global__ __launch_bounds__(256) void gemm_xwT(const bf16_t* __restrict__ X,
                                                const bf16_t* __restrict__ W,
                                                void* __restrict__ outp,
                                                const int* __restrict__ segp) {
  __shared__ __align__(16) bf16_t lA[128 * 72];
  __shared__ __align__(16) bf16_t lB[256 * 72];
  const int tid   = threadIdx.x;
  const int lane  = tid & 31, wave = tid >> 5;
  const int wm    = wave >> 2, wn = wave & 3;      // 2x4 wave grid
  const int lr    = lane & 15, lhalf = lane >> 4;
  const long Mb   = (long)blockIdx.y * 128;
  const long Nb   = (long)blockIdx.x * 256;
  const int  ldr  = tid >> 1;
  const int  ldc  = (tid & 1) * 32;
  const bf16_t* gA  = X + (Mb + ldr) * HID + ldc;
  const bf16_t* gB0 = W + (Nb + ldr) * HID + ldc;
  const bf16_t* gB1 = W + (Nb + 128 + ldr) * HID + ldc;
  bf16_t* sA  = lA + ldr * 72 + ldc;
  bf16_t* sB0 = lB + ldr * 72 + ldc;
  bf16_t* sB1 = lB + (128 + ldr) * 72 + ldc;

  v8f acc[4][4] = {};
  for (int kt = 0; kt < HID; kt += 64) {
    __syncthreads();
#if USE_ASYNC
#pragma unroll
    for (int c = 0; c < 4; ++c) {
      async_b128(gA + kt + c * 8, sA + c * 8);
      async_b128(gB0 + kt + c * 8, sB0 + c * 8);
      async_b128(gB1 + kt + c * 8, sB1 + c * 8);
    }
    async_wait();
#else
#pragma unroll
    for (int c = 0; c < 4; ++c) {
      *(uint4*)(sA + c * 8)  = *(const uint4*)(gA + kt + c * 8);
      *(uint4*)(sB0 + c * 8) = *(const uint4*)(gB0 + kt + c * 8);
      *(uint4*)(sB1 + c * 8) = *(const uint4*)(gB1 + kt + c * 8);
    }
#endif
    __syncthreads();
#pragma unroll
    for (int ks = 0; ks < 2; ++ks) {
      v16bf a[4], b[4];
#pragma unroll
      for (int i = 0; i < 4; ++i)
        a[i] = frag_row(lA + (wm * 64 + i * 16 + lr) * 72 + ks * 32, lhalf);
#pragma unroll
      for (int j = 0; j < 4; ++j)
        b[j] = frag_row(lB + (wn * 64 + j * 16 + lr) * 72 + ks * 32, lhalf);
#pragma unroll
      for (int i = 0; i < 4; ++i)
#pragma unroll
        for (int j = 0; j < 4; ++j)
          acc[i][j] = __builtin_amdgcn_wmma_f32_16x16x32_bf16(
              false, a[i], false, b[j], (short)0, acc[i][j], false, false);
    }
  }

  const int seg = (MODE == 1) ? segp[0] : 0;
#pragma unroll
  for (int i = 0; i < 4; ++i)
#pragma unroll
    for (int j = 0; j < 4; ++j)
#pragma unroll
      for (int r = 0; r < 8; ++r) {
        const long row = Mb + wm * 64 + i * 16 + lhalf * 8 + r;
        const long col = Nb + wn * 64 + j * 16 + lr;
        const float v = acc[i][j][r];
        if constexpr (MODE == 0) {
          __builtin_nontemporal_store((bf16_t)elu1(v), (bf16_t*)outp + row * HID + col);
        } else if constexpr (MODE == 1) {
          const int   h     = (int)(col >> 7);
          const float slope = exp2f(-0.5f * (float)(h + 1));
          const float wh    = __expf(-slope * (float)seg);
          __builtin_nontemporal_store((bf16_t)(elu1(v) * wh), (bf16_t*)outp + row * HID + col);
        } else if constexpr (MODE == 2) {
          __builtin_nontemporal_store((bf16_t)v, (bf16_t*)outp + row * HID + col);
        } else {
          ((float*)outp)[row * HID + col] = v;
        }
      }
}

// ---------------------------------------------------------------------------
// Per-head retrieval: out[s,e] = sum_d A[s,d] * memory[h,d,e], plus row norm.
// memory[h] (128x128 f32, 64KB) is fetched by the Tensor Data Mover into an
// LDS staging buffer (one 2-D D# descriptor), then converted/transposed.
// MODE 0: attn = out/norm (bf16).  MODE 1: delta_v = v - out/norm (bf16).
// ---------------------------------------------------------------------------
template <int MODE>
__global__ __launch_bounds__(256) void retrieve_head(const bf16_t* __restrict__ A,
                                                     const float* __restrict__ mem,
                                                     const float* __restrict__ memnorm,
                                                     const bf16_t* __restrict__ Vb,
                                                     bf16_t* __restrict__ outb) {
  __shared__ __align__(16) bf16_t lsq[128 * 136];
  __shared__ __align__(16) bf16_t lmt[128 * 136];   // memory transposed [e][d]
  __shared__ float lnorm[128];
  __shared__ float lrown[128];
  const int tid  = threadIdx.x;
  const int lane = tid & 31, wave = tid >> 5;
  const int wm   = wave >> 1, wn = wave & 1;
  const int lr   = lane & 15, lhalf = lane >> 4;
  const int bh   = blockIdx.y;
  const int bidx = bh >> 4, h = bh & 15;
  const long mrow0 = (long)bidx * SEQ + (long)blockIdx.x * 128;
  const float* gm = mem + (long)h * DHEAD * DHEAD;

  {  // stage sigma tile [128 s x 128 d]
    const int row = tid >> 1, cb = (tid & 1) * 64;
    const bf16_t* g = A + (mrow0 + row) * HID + h * DHEAD + cb;
    bf16_t* s = lsq + row * 136 + cb;
#if USE_ASYNC
#pragma unroll
    for (int c = 0; c < 8; ++c) async_b128(g + c * 8, s + c * 8);
#else
#pragma unroll
    for (int c = 0; c < 8; ++c) *(uint4*)(s + c * 8) = *(const uint4*)(g + c * 8);
#endif
  }

#if USE_TDM
  __shared__ __align__(16) float stageF[DHEAD * DHEAD];
  if (wave == 0) {
    // D# group 0: count=1 | lds_addr | 57-bit global addr | type=2
    const unsigned long long ga = (unsigned long long)(size_t)gm;
    const unsigned lo = (unsigned)(size_t)(lds_char*)&stageF[0];
    u32x4 g0;
    g0[0] = 1u;
    g0[1] = lo;
    g0[2] = (unsigned)(ga & 0xFFFFFFFFull);
    g0[3] = (unsigned)((ga >> 32) & 0x01FFFFFFull) | (2u << 30);
    // D# group 1: data_size=4B; tensor 128x128, tile 128x128, stride 128
    i32x8 g1 = {};
    g1[0] = 2 << 16;        // data_size=2 (4 bytes), wg_mask=0
    g1[1] = 128 << 16;      // tensor_dim0[15:0] in bits[63:48]
    g1[2] = 128 << 16;      // tensor_dim1[15:0] in bits[95:80]
    g1[3] = 128 << 16;      // tile_dim0 in bits[127:112]
    g1[4] = 128;            // tile_dim1 in bits[143:128]
    g1[5] = 128;            // tensor_dim0_stride low in bits[191:160]
    const i32x4 gz = {};
#if __clang_major__ >= 23
    const i32x8 gz8 = {};
    __builtin_amdgcn_tensor_load_to_lds(g0, g1, gz, gz, gz8, 0);
#else
    __builtin_amdgcn_tensor_load_to_lds(g0, g1, gz, gz, 0);
#endif
    __builtin_amdgcn_s_wait_tensorcnt(0);
  }
  __syncthreads();
  {  // convert + transpose from LDS staging
    for (int it = 0; it < 16; ++it) {
      const int idx = (tid + it * 256) * 4;
      const int d = idx >> 7, e = idx & 127;
      const float4 f = *(const float4*)(stageF + idx);
      lmt[(e + 0) * 136 + d] = (bf16_t)f.x;
      lmt[(e + 1) * 136 + d] = (bf16_t)f.y;
      lmt[(e + 2) * 136 + d] = (bf16_t)f.z;
      lmt[(e + 3) * 136 + d] = (bf16_t)f.w;
    }
    if (tid < 128) lnorm[tid] = memnorm[h * DHEAD + tid];
  }
#else
  {  // stage memory[h] transposed, as bf16 (vectorized f32 reads, u16 scatter)
    for (int it = 0; it < 16; ++it) {
      const int idx = (tid + it * 256) * 4;
      const int d = idx >> 7, e = idx & 127;
      const float4 f = *(const float4*)(gm + idx);
      lmt[(e + 0) * 136 + d] = (bf16_t)f.x;
      lmt[(e + 1) * 136 + d] = (bf16_t)f.y;
      lmt[(e + 2) * 136 + d] = (bf16_t)f.z;
      lmt[(e + 3) * 136 + d] = (bf16_t)f.w;
    }
    if (tid < 128) lnorm[tid] = memnorm[h * DHEAD + tid];
  }
#endif
#if USE_ASYNC
  async_wait();
#endif
  __syncthreads();

  v8f acc[2][4] = {};
#pragma unroll
  for (int ks = 0; ks < 4; ++ks) {
    v16bf a[2], b[4];
#pragma unroll
    for (int i = 0; i < 2; ++i)
      a[i] = frag_row(lsq + (wm * 32 + i * 16 + lr) * 136 + ks * 32, lhalf);
#pragma unroll
    for (int j = 0; j < 4; ++j)
      b[j] = frag_row(lmt + (wn * 64 + j * 16 + lr) * 136 + ks * 32, lhalf);
#pragma unroll
    for (int i = 0; i < 2; ++i)
#pragma unroll
      for (int j = 0; j < 4; ++j)
        acc[i][j] = __builtin_amdgcn_wmma_f32_16x16x32_bf16(
            false, a[i], false, b[j], (short)0, acc[i][j], false, false);
  }

  if (tid < 128) {
    float n = 0.0f;
#pragma unroll
    for (int c = 0; c < 16; ++c) {
      const v8bf ch = *(const v8bf*)(lsq + tid * 136 + c * 8);
#pragma unroll
      for (int i = 0; i < 8; ++i) n += (float)ch[i] * lnorm[c * 8 + i];
    }
    lrown[tid] = fmaxf(n, 1e-6f);
  }
  __syncthreads();

#pragma unroll
  for (int i = 0; i < 2; ++i)
#pragma unroll
    for (int j = 0; j < 4; ++j)
#pragma unroll
      for (int r = 0; r < 8; ++r) {
        const int srow = wm * 32 + i * 16 + lhalf * 8 + r;
        const int e    = wn * 64 + j * 16 + lr;
        const float val = acc[i][j][r] / lrown[srow];
        const long gi = (mrow0 + srow) * HID + h * DHEAD + e;
        if constexpr (MODE == 0) {
          __builtin_nontemporal_store((bf16_t)val, outb + gi);
        } else {
          __builtin_nontemporal_store((bf16_t)((float)Vb[gi] - val), outb + gi);
        }
      }
}

// ---------------------------------------------------------------------------
// new_memory[h,d,e] += (1/(B*S)) * sum_m wk[m,h,d] * dv[m,h,e]  (split-K atomics)
// new_memory_norm[h,d] += (1/B) * sum_m wk[m,h,d]
// Tiles are transposed at LDS-fill time so fragments are vectorized frag_row.
// ---------------------------------------------------------------------------
__global__ __launch_bounds__(256) void update_memory(const bf16_t* __restrict__ WK,
                                                     const bf16_t* __restrict__ DV,
                                                     float* __restrict__ outMem,
                                                     float* __restrict__ outNorm) {
  __shared__ __align__(16) bf16_t lwkT[128 * 48];  // [d][m]
  __shared__ __align__(16) bf16_t ldvT[128 * 48];  // [e][m]
  const int tid  = threadIdx.x;
  const int lane = tid & 31, wave = tid >> 5;
  const int wm   = wave >> 1, wn = wave & 1;
  const int lr   = lane & 15, lhalf = lane >> 4;
  const int h    = blockIdx.y;
  const long m0  = (long)blockIdx.x * 2048;
  const int row  = tid >> 3, cb = (tid & 7) * 16;

  v8f acc[2][4] = {};
  float colsum = 0.0f;
  for (int mt = 0; mt < 2048; mt += 32) {
    __syncthreads();
    const long g = (m0 + mt + row) * HID + h * DHEAD + cb;
    const v8bf w0 = *(const v8bf*)(WK + g);
    const v8bf w1 = *(const v8bf*)(WK + g + 8);
    const v8bf d0 = *(const v8bf*)(DV + g);
    const v8bf d1 = *(const v8bf*)(DV + g + 8);
#pragma unroll
    for (int i = 0; i < 8; ++i) {
      lwkT[(cb + i) * 48 + row]     = w0[i];
      lwkT[(cb + 8 + i) * 48 + row] = w1[i];
      ldvT[(cb + i) * 48 + row]     = d0[i];
      ldvT[(cb + 8 + i) * 48 + row] = d1[i];
    }
    __syncthreads();
    if (tid < 128) {
      float cs = 0.0f;
#pragma unroll
      for (int c = 0; c < 4; ++c) {
        const v8bf ch = *(const v8bf*)(lwkT + tid * 48 + c * 8);
#pragma unroll
        for (int i = 0; i < 8; ++i) cs += (float)ch[i];
      }
      colsum += cs;
    }
    v16bf a[2], b[4];
#pragma unroll
    for (int i = 0; i < 2; ++i)
      a[i] = frag_row(lwkT + (wm * 32 + i * 16 + lr) * 48, lhalf);
#pragma unroll
    for (int j = 0; j < 4; ++j)
      b[j] = frag_row(ldvT + (wn * 64 + j * 16 + lr) * 48, lhalf);
#pragma unroll
    for (int i = 0; i < 2; ++i)
#pragma unroll
      for (int j = 0; j < 4; ++j)
        acc[i][j] = __builtin_amdgcn_wmma_f32_16x16x32_bf16(
            false, a[i], false, b[j], (short)0, acc[i][j], false, false);
  }

  const float scale = 1.0f / (float)MROWS;
#pragma unroll
  for (int i = 0; i < 2; ++i)
#pragma unroll
    for (int j = 0; j < 4; ++j)
#pragma unroll
      for (int r = 0; r < 8; ++r) {
        const int d = wm * 32 + i * 16 + lhalf * 8 + r;
        const int e = wn * 64 + j * 16 + lr;
        atomicAdd(outMem + (long)h * (DHEAD * DHEAD) + d * DHEAD + e, acc[i][j][r] * scale);
      }
  if (tid < 128) atomicAdd(outNorm + h * DHEAD + tid, colsum * 0.25f);
}

// ---------------------------------------------------------------------------
__global__ void f32_to_bf16_k(const float* __restrict__ s, bf16_t* __restrict__ d, int n) {
  const int stride = blockDim.x * gridDim.x * 4;
  for (int i = (blockIdx.x * blockDim.x + threadIdx.x) * 4; i < n; i += stride) {
    const float4 f = *(const float4*)(s + i);
    v4bf o;
    o[0] = (bf16_t)f.x; o[1] = (bf16_t)f.y; o[2] = (bf16_t)f.z; o[3] = (bf16_t)f.w;
    __builtin_nontemporal_store(o, (v4bf*)(d + i));
  }
}

__global__ void init_tail_k(const float* __restrict__ mem, const float* __restrict__ memnorm,
                            float* __restrict__ outMem, float* __restrict__ outNorm) {
  const int i = blockIdx.x * blockDim.x + threadIdx.x;
  if (i < NHEAD * DHEAD * DHEAD) outMem[i] = mem[i];
  if (i < NHEAD * DHEAD) outNorm[i] = memnorm[i];
}

// ---------------------------------------------------------------------------
extern "C" void kernel_launch(void* const* d_in, const int* in_sizes, int n_in,
                              void* d_out, int out_size, void* d_ws, size_t ws_size,
                              hipStream_t stream) {
  (void)in_sizes; (void)n_in; (void)out_size; (void)ws_size;
  const float* hs   = (const float*)d_in[0];
  const float* wq   = (const float*)d_in[1];
  const float* wk   = (const float*)d_in[2];
  const float* wv   = (const float*)d_in[3];
  const float* wo   = (const float*)d_in[4];
  const float* mem  = (const float*)d_in[5];
  const float* mnrm = (const float*)d_in[6];
  const int*   seg  = (const int*)d_in[7];

  float* out     = (float*)d_out;                      // [M, HID]
  float* outMem  = out + (long)MROWS * HID;            // [16,128,128]
  float* outNorm = outMem + NHEAD * DHEAD * DHEAD;     // [16,128]

  char* p = (char*)d_ws;
  auto alloc = [&](size_t bytes) { char* r = p; p += (bytes + 255) & ~(size_t)255; return r; };
  bf16_t* Xb  = (bf16_t*)alloc((size_t)MROWS * HID * 2);
  bf16_t* Wqb = (bf16_t*)alloc((size_t)HID * HID * 2);
  bf16_t* Wkb = (bf16_t*)alloc((size_t)HID * HID * 2);
  bf16_t* Wvb = (bf16_t*)alloc((size_t)HID * HID * 2);
  bf16_t* Wob = (bf16_t*)alloc((size_t)HID * HID * 2);
  bf16_t* SQ  = (bf16_t*)alloc((size_t)MROWS * HID * 2);
  bf16_t* WKw = (bf16_t*)alloc((size_t)MROWS * HID * 2);
  bf16_t* Vb  = (bf16_t*)alloc((size_t)MROWS * HID * 2);
  bf16_t* ATT = (bf16_t*)alloc((size_t)MROWS * HID * 2);
  bf16_t* DV  = (bf16_t*)alloc((size_t)MROWS * HID * 2);

  const dim3 blk(256);
  f32_to_bf16_k<<<4096, blk, 0, stream>>>(hs, Xb, MROWS * HID);
  f32_to_bf16_k<<<1024, blk, 0, stream>>>(wq, Wqb, HID * HID);
  f32_to_bf16_k<<<1024, blk, 0, stream>>>(wk, Wkb, HID * HID);
  f32_to_bf16_k<<<1024, blk, 0, stream>>>(wv, Wvb, HID * HID);
  f32_to_bf16_k<<<1024, blk, 0, stream>>>(wo, Wob, HID * HID);

  const dim3 ggrid(HID / 256, MROWS / 128);
  gemm_xwT<0><<<ggrid, blk, 0, stream>>>(Xb, Wqb, SQ, seg);
  gemm_xwT<1><<<ggrid, blk, 0, stream>>>(Xb, Wkb, WKw, seg);
  gemm_xwT<2><<<ggrid, blk, 0, stream>>>(Xb, Wvb, Vb, seg);

  const dim3 rgrid(SEQ / 128, BATCH * NHEAD);
  retrieve_head<0><<<rgrid, blk, 0, stream>>>(SQ, mem, mnrm, nullptr, ATT);
  retrieve_head<1><<<rgrid, blk, 0, stream>>>(WKw, mem, mnrm, Vb, DV);

  init_tail_k<<<(NHEAD * DHEAD * DHEAD + 255) / 256, blk, 0, stream>>>(mem, mnrm, outMem, outNorm);

  const dim3 ugrid(MROWS / 2048, NHEAD);
  update_memory<<<ugrid, blk, 0, stream>>>(WKw, DV, outMem, outNorm);

  gemm_xwT<3><<<ggrid, blk, 0, stream>>>(ATT, Wob, out, seg);
}
